// LANDER_39006892982821
// MI455X (gfx1250) — matrix operand order repaired and checked by reading, not compiled, not run on hardware
//
#include <hip/hip_runtime.h>
#include <hip/hip_bf16.h>

typedef __attribute__((ext_vector_type(16))) _Float16 v16h;
typedef __attribute__((ext_vector_type(8)))  float    v8f;

#define DD 64

// ---------------------------------------------------------------------------
// WMMA fragment layouts (CDNA5 ISA §7.12.2, wave32):
// A 16x32 f16: lanes 0-15 (M=lane): elems 0..7 -> K=0..7, 8..15 -> K=16..23;
//              lanes 16-31 (M=lane-16): elems 0..7 -> K=8..15, 8..15 -> K=24..31
//   inverse for local K: kh=(K>>3)&1, e=(K&7)+((K>>4)<<3), lane=kh*16+m
// B 32x16 f16: lanes 0-15 (N=lane): elems 0..7 -> K=0..7, 8..15 -> K=16..23; lanes 16-31: K+8
// C 16x16 f32: lanes 0-15 (N=lane): VGPR r -> M=r; lanes 16-31: M=r+8
// Operands are pre-swizzled in LDS so each lane reads 16 contiguous f16
// (ds_load_b128 x2). fp32 accuracy recovered via split-f16 compensation:
//   x = hi + lo,  A*B ~= Ahi*Bhi + Ahi*Blo + Alo*Bhi   (3 WMMAs / tile-step)
// ---------------------------------------------------------------------------

__device__ __forceinline__ void splitf(float x, _Float16& hi, _Float16& lo) {
    _Float16 h = (_Float16)x;
    hi = h;
    lo = (_Float16)(x - (float)h);
}

// Node transform: out = feat @ W + bias, [N x 64] @ [64 x 64].
__global__ __launch_bounds__(256) void lander_node_gemm(
    const float* __restrict__ feat_src, const float* __restrict__ feat_dst,
    const float* __restrict__ Wsrc, const float* __restrict__ bsrc,
    const float* __restrict__ Wdst, const float* __restrict__ bdst,
    float* __restrict__ ts, float* __restrict__ td, int N)
{
    const float* feat; const float* W; const float* bias; float* out;
    if (blockIdx.y == 0) { feat = feat_src; W = Wsrc; bias = bsrc; out = ts; }
    else                 { feat = feat_dst; W = Wdst; bias = bdst; out = td; }

    __shared__ __align__(32) _Float16 sWhi[4 * 2 * 32 * 16]; // [ntile][kk][lane][e]
    __shared__ __align__(32) _Float16 sWlo[4 * 2 * 32 * 16];
    __shared__ __align__(32) _Float16 sAhi[8 * 2 * 32 * 16]; // [tile][kk][lane][e]
    __shared__ __align__(32) _Float16 sAlo[8 * 2 * 32 * 16];
    __shared__ float sBias[DD];

    const int t = threadIdx.x;

    // Swizzle W (f32 row-major [K][N]) into B-fragment layout, split f16.
    #pragma unroll
    for (int i = t; i < 4096; i += 256) {
        int e = i & 15, lane = (i >> 4) & 31, kk = (i >> 9) & 1, nt = i >> 10;
        int n  = nt * 16 + (lane & 15);
        int kh = lane >> 4;
        int k  = kk * 32 + kh * 8 + (e & 7) + ((e >> 3) << 4);
        _Float16 h, l; splitf(W[k * DD + n], h, l);
        sWhi[i] = h; sWlo[i] = l;
    }
    if (t < DD) sBias[t] = bias[t];

    // Stage 128 feature rows into A-fragment layout, split f16.
    const int base_row = blockIdx.x * 128;
    {
        int le = t >> 1;
        int cb = (t & 1) * 32;
        int row = base_row + le;
        int srow = row < N ? row : (N - 1);
        const float4* rp = (const float4*)(feat + (long long)srow * DD + cb);
        int m = le & 15, tile = le >> 4;
        #pragma unroll
        for (int j = 0; j < 8; ++j) {
            float4 v = rp[j];
            float vv[4] = {v.x, v.y, v.z, v.w};
            #pragma unroll
            for (int q = 0; q < 4; ++q) {
                int c  = cb + j * 4 + q;
                int kk = c >> 5, kl = c & 31;
                int kh = (kl >> 3) & 1;
                int e  = (kl & 7) + ((kl >> 4) << 3);
                int lane = kh * 16 + m;
                int idx = ((tile * 2 + kk) * 32 + lane) * 16 + e;
                _Float16 h, l; splitf(vv[q], h, l);
                sAhi[idx] = h; sAlo[idx] = l;
            }
        }
    }
    __syncthreads();

    const int wave = t >> 5;
    const int lane = t & 31;
    v8f acc[4];
    v8f z = {0.f, 0.f, 0.f, 0.f, 0.f, 0.f, 0.f, 0.f};
    #pragma unroll
    for (int i = 0; i < 4; ++i) acc[i] = z;

    #pragma unroll
    for (int kk = 0; kk < 2; ++kk) {
        int aidx = ((wave * 2 + kk) * 32 + lane) * 16;
        v16h ah = *(const v16h*)&sAhi[aidx];
        v16h al = *(const v16h*)&sAlo[aidx];
        #pragma unroll
        for (int nt = 0; nt < 4; ++nt) {
            int bidx = ((nt * 2 + kk) * 32 + lane) * 16;
            v16h bh = *(const v16h*)&sWhi[bidx];
            v16h bl = *(const v16h*)&sWlo[bidx];
            acc[nt] = __builtin_amdgcn_wmma_f32_16x16x32_f16(
                false, ah, false, bh, (short)0, acc[nt], false, false);
            acc[nt] = __builtin_amdgcn_wmma_f32_16x16x32_f16(
                false, ah, false, bl, (short)0, acc[nt], false, false);
            acc[nt] = __builtin_amdgcn_wmma_f32_16x16x32_f16(
                false, al, false, bh, (short)0, acc[nt], false, false);
        }
    }

    const int coln = lane & 15;
    const int rhi  = (lane >> 4) * 8;
    if (base_row + 128 <= N) {   // full-tile fast path: no per-element guards
        #pragma unroll
        for (int nt = 0; nt < 4; ++nt) {
            int col = nt * 16 + coln;
            float bv = sBias[col];
            #pragma unroll
            for (int r = 0; r < 8; ++r) {
                int row = base_row + wave * 16 + rhi + r;
                out[(long long)row * DD + col] = acc[nt][r] + bv;
            }
        }
    } else {
        #pragma unroll
        for (int nt = 0; nt < 4; ++nt) {
            int col = nt * 16 + coln;
            float bv = sBias[col];
            #pragma unroll
            for (int r = 0; r < 8; ++r) {
                int row = base_row + wave * 16 + rhi + r;
                if (row < N) out[(long long)row * DD + col] = acc[nt][r] + bv;
            }
        }
    }
}

__global__ __launch_bounds__(256) void lander_zero(float* __restrict__ p, int n) {
    int i = blockIdx.x * 256 + threadIdx.x;
    if (i < n) p[i] = 0.f;
}

// Per-edge: h1 = prelu(ts[src]+td[dst], a1); h2 = prelu(h1@W1+b1, a2);
// pred = h2@W2+b2; softmax; scatter msg / count.
__global__ __launch_bounds__(256) void lander_edge(
    const float* __restrict__ ts, const float* __restrict__ td,
    const int* __restrict__ src_ids, const int* __restrict__ dst_ids,
    const float* __restrict__ raw_affine,
    const float* __restrict__ alpha1, const float* __restrict__ W1,
    const float* __restrict__ b1, const float* __restrict__ alpha2,
    const float* __restrict__ W2, const float* __restrict__ b2,
    float* __restrict__ pred_conn, float* __restrict__ prob_conn,
    float* __restrict__ den_sum, float* __restrict__ cnt, int E)
{
    __shared__ __align__(32) _Float16 sWhi[4 * 2 * 32 * 16];
    __shared__ __align__(32) _Float16 sWlo[4 * 2 * 32 * 16];
    __shared__ __align__(32) _Float16 sAhi[8 * 2 * 32 * 16];
    __shared__ __align__(32) _Float16 sAlo[8 * 2 * 32 * 16];
    __shared__ float sW2[DD * 2];
    __shared__ float sB1[DD], sA2[DD];
    __shared__ float sB2[2];

    const int t = threadIdx.x;
    const long long base = (long long)blockIdx.x * 128;

    // Prefetch next tile's edge ids into L2 (global_prefetch_b8).
    if ((long long)(blockIdx.x + 1) * 128 < E) {
        if (t < 4)       __builtin_prefetch(src_ids + base + 128 + t * 32, 0, 1);
        else if (t < 8)  __builtin_prefetch(dst_ids + base + 128 + (t - 4) * 32, 0, 1);
    }

    #pragma unroll
    for (int i = t; i < 4096; i += 256) {
        int e = i & 15, lane = (i >> 4) & 31, kk = (i >> 9) & 1, nt = i >> 10;
        int n  = nt * 16 + (lane & 15);
        int kh = lane >> 4;
        int k  = kk * 32 + kh * 8 + (e & 7) + ((e >> 3) << 4);
        _Float16 h, l; splitf(W1[k * DD + n], h, l);
        sWhi[i] = h; sWlo[i] = l;
    }
    if (t < DD) {
        sB1[t] = b1[t]; sA2[t] = alpha2[t];
        sW2[2 * t] = W2[2 * t]; sW2[2 * t + 1] = W2[2 * t + 1];
    }
    if (t < 2) sB2[t] = b2[t];

    // Gather ts[src]+td[dst], PReLU(alpha1), split-f16 into A fragments.
    {
        int le = t >> 1;
        int cb = (t & 1) * 32;
        long long e = base + le;
        long long ec = e < E ? e : (long long)(E - 1);
        int s = src_ids[ec], d = dst_ids[ec];
        const float4* sp = (const float4*)(ts + (long long)s * DD + cb);
        const float4* dp = (const float4*)(td + (long long)d * DD + cb);
        int m = le & 15, tile = le >> 4;
        #pragma unroll
        for (int j = 0; j < 8; ++j) {
            float4 va = sp[j], vb = dp[j];
            float vv[4] = {va.x + vb.x, va.y + vb.y, va.z + vb.z, va.w + vb.w};
            #pragma unroll
            for (int q = 0; q < 4; ++q) {
                int c = cb + j * 4 + q;
                float al = alpha1[c];
                float h = vv[q];
                h = h > 0.f ? h : al * h;
                int kk = c >> 5, kl = c & 31;
                int kh = (kl >> 3) & 1;
                int ee = (kl & 7) + ((kl >> 4) << 3);
                int lane = kh * 16 + m;
                int idx = ((tile * 2 + kk) * 32 + lane) * 16 + ee;
                _Float16 hh, ll; splitf(h, hh, ll);
                sAhi[idx] = hh; sAlo[idx] = ll;
            }
        }
    }
    __syncthreads();

    const int wave = t >> 5;
    const int lane = t & 31;
    v8f acc[4];
    v8f z = {0.f, 0.f, 0.f, 0.f, 0.f, 0.f, 0.f, 0.f};
    #pragma unroll
    for (int i = 0; i < 4; ++i) acc[i] = z;

    #pragma unroll
    for (int kk = 0; kk < 2; ++kk) {
        int aidx = ((wave * 2 + kk) * 32 + lane) * 16;
        v16h ah = *(const v16h*)&sAhi[aidx];
        v16h al = *(const v16h*)&sAlo[aidx];
        #pragma unroll
        for (int nt = 0; nt < 4; ++nt) {
            int bidx = ((nt * 2 + kk) * 32 + lane) * 16;
            v16h bh = *(const v16h*)&sWhi[bidx];
            v16h bl = *(const v16h*)&sWlo[bidx];
            acc[nt] = __builtin_amdgcn_wmma_f32_16x16x32_f16(
                false, ah, false, bh, (short)0, acc[nt], false, false);
            acc[nt] = __builtin_amdgcn_wmma_f32_16x16x32_f16(
                false, ah, false, bl, (short)0, acc[nt], false, false);
            acc[nt] = __builtin_amdgcn_wmma_f32_16x16x32_f16(
                false, al, false, bh, (short)0, acc[nt], false, false);
        }
    }

    // W2 head folded into registers: per-lane partial dots over this lane's
    // 4 columns, then 16-lane xor-tree reduction (columns of each row are
    // spread across lanes 0..15 / 16..31 respectively).
    const int coln = lane & 15;
    const int rhi  = (lane >> 4) * 8;
    float p0[8], p1[8];
    #pragma unroll
    for (int r = 0; r < 8; ++r) { p0[r] = 0.f; p1[r] = 0.f; }
    #pragma unroll
    for (int nt = 0; nt < 4; ++nt) {
        int col = nt * 16 + coln;
        float bv = sB1[col], av = sA2[col];
        float w0 = sW2[2 * col], w1 = sW2[2 * col + 1];
        #pragma unroll
        for (int r = 0; r < 8; ++r) {
            float v = acc[nt][r] + bv;
            v = v > 0.f ? v : av * v;   // prelu(alpha2)
            p0[r] += v * w0;
            p1[r] += v * w1;
        }
    }
    #pragma unroll
    for (int msk = 1; msk < 16; msk <<= 1) {
        #pragma unroll
        for (int r = 0; r < 8; ++r) {
            p0[r] += __shfl_xor(p0[r], msk, 32);
            p1[r] += __shfl_xor(p1[r], msk, 32);
        }
    }

    const int sub = lane & 15;
    if (sub < 8) {
        long long e = base + wave * 16 + rhi + sub;
        if (e < E) {
            float z0 = p0[sub] + sB2[0];
            float z1 = p1[sub] + sB2[1];
            float2 pc = {z0, z1};
            *(float2*)&pred_conn[e * 2] = pc;
            float mx = fmaxf(z0, z1);
            float e0 = __expf(z0 - mx), e1 = __expf(z1 - mx);
            float inv = 1.f / (e0 + e1);
            float2 pb = {e0 * inv, e1 * inv};
            *(float2*)&prob_conn[e * 2] = pb;
            int d = dst_ids[e];
            float msg = raw_affine[e] * (pb.y - pb.x);
            atomicAdd(&den_sum[d], msg);
            atomicAdd(&cnt[d], 1.0f);
        }
    }
}

__global__ __launch_bounds__(256) void lander_final(
    const float* __restrict__ den, const float* __restrict__ cnt,
    float* __restrict__ pred_den, int N)
{
    int i = blockIdx.x * 256 + threadIdx.x;
    if (i < N) pred_den[i] = den[i] / fmaxf(cnt[i], 1.0f);
}

extern "C" void kernel_launch(void* const* d_in, const int* in_sizes, int n_in,
                              void* d_out, int out_size, void* d_ws, size_t ws_size,
                              hipStream_t stream) {
    const float* feat_src   = (const float*)d_in[0];
    const float* feat_dst   = (const float*)d_in[1];
    const float* raw_affine = (const float*)d_in[2];
    const int*   src_ids    = (const int*)d_in[3];
    const int*   dst_ids    = (const int*)d_in[4];
    const float* Wsrc       = (const float*)d_in[5];
    const float* bsrc       = (const float*)d_in[6];
    const float* Wdst       = (const float*)d_in[7];
    const float* bdst       = (const float*)d_in[8];
    const float* alpha1     = (const float*)d_in[9];
    const float* W1         = (const float*)d_in[10];
    const float* b1         = (const float*)d_in[11];
    const float* alpha2     = (const float*)d_in[12];
    const float* W2         = (const float*)d_in[13];
    const float* b2         = (const float*)d_in[14];

    const int N = in_sizes[0] / DD;   // 100000
    const int E = in_sizes[2];        // 1200000

    char* ws = (char*)d_ws;
    float* ts  = (float*)ws;
    float* td  = (float*)(ws + (size_t)N * DD * sizeof(float));
    float* den = (float*)(ws + (size_t)2 * N * DD * sizeof(float));
    float* cnt = den + N;

    float* pred_conn = (float*)d_out;
    float* prob_conn = pred_conn + (size_t)E * 2;
    float* pred_den  = pred_conn + (size_t)E * 4;

    lander_zero<<<(2 * N + 255) / 256, 256, 0, stream>>>(den, 2 * N);

    dim3 gA((N + 127) / 128, 2, 1);
    lander_node_gemm<<<gA, 256, 0, stream>>>(feat_src, feat_dst, Wsrc, bsrc,
                                             Wdst, bdst, ts, td, N);

    lander_edge<<<(E + 127) / 128, 256, 0, stream>>>(
        ts, td, src_ids, dst_ids, raw_affine, alpha1, W1, b1, alpha2, W2, b2,
        pred_conn, prob_conn, den, cnt, E);

    lander_final<<<(N + 255) / 256, 256, 0, stream>>>(den, cnt, pred_den, N);
}
